// BlockSparseRingMultiheadDilatedAttention_86354612453508
// MI455X (gfx1250) — matrix operand order repaired and checked by reading, not compiled, not run on hardware
//
#include <hip/hip_runtime.h>
#include <hip/hip_bf16.h>

// ---------------------------------------------------------------------------
// Types for CDNA5 WMMA (wave32): V_WMMA_F32_16X16X32_BF16
// ---------------------------------------------------------------------------
typedef __attribute__((ext_vector_type(16))) __bf16 bf16x16;
typedef __attribute__((ext_vector_type(8)))  float  f32x8;

union BFrag {
    bf16x16      v;
    unsigned int u[8];
    uint4        q[2];
};

#define EMBED_DIM 768
#define SEQ_LEN   8192

// Workspace layout (bytes). Gathered rows per head: r=1 -> 8192, r=2 -> 4096,
// r=4 -> 2048. Total gathered rows = 4*8192 + 4*4096 + 4*2048 = 57344.
#define QG_BYTES    (57344u * 64u * 2u)              // 7,340,032
#define KG_OFF      QG_BYTES
#define VT_OFF      (2u * QG_BYTES)
#define ATTN_OFF    (3u * QG_BYTES)                  // 22,020,096
#define ATTN_BYTES  (SEQ_LEN * EMBED_DIM * 2u)       // 12,582,912
#define XB_OFF      (ATTN_OFF + ATTN_BYTES)          // 34,603,008
#define XB_BYTES    (SEQ_LEN * EMBED_DIM * 2u)
#define WQKV_OFF    (XB_OFF + XB_BYTES)              // 47,185,920
#define WQKV_BYTES  (3u * EMBED_DIM * EMBED_DIM * 2u)
#define WOUT_OFF    (WQKV_OFF + WQKV_BYTES)          // 50,724,864

__device__ __forceinline__ int head_base_rows(int h) {
    if (h < 4) return h * 8192;
    if (h < 8) return 32768 + (h - 4) * 4096;
    return 49152 + (h - 8) * 2048;
}

__device__ __forceinline__ unsigned int pack_bf16_2(float lo, float hi) {
    unsigned a = __float_as_uint(lo), b = __float_as_uint(hi);
    unsigned ra = (a + 0x7FFFu + ((a >> 16) & 1u)) >> 16;
    unsigned rb = (b + 0x7FFFu + ((b >> 16) & 1u)) >> 16;
    return (ra & 0xFFFFu) | (rb << 16);
}

__device__ __forceinline__ unsigned short f32_to_bf16u(float f) {
    unsigned a = __float_as_uint(f);
    return (unsigned short)((a + 0x7FFFu + ((a >> 16) & 1u)) >> 16);
}

// ---------------------------------------------------------------------------
// Kernel A: one-shot fp32 -> packed bf16 conversion (streaming, BW-bound).
// ---------------------------------------------------------------------------
__global__ void cvt_f32_bf16(const float4* __restrict__ src,
                             uint2* __restrict__ dst, int n4) {
    int i = blockIdx.x * blockDim.x + threadIdx.x;
    for (; i < n4; i += gridDim.x * blockDim.x) {
        float4 f = src[i];
        uint2 o;
        o.x = pack_bf16_2(f.x, f.y);
        o.y = pack_bf16_2(f.z, f.w);
        dst[i] = o;
    }
}

// ---------------------------------------------------------------------------
// Kernel 0: zero the bf16 attention buffer (dilated heads leave holes = 0)
// ---------------------------------------------------------------------------
__global__ void zero_attn_kernel(uint4* __restrict__ p, int n4) {
    int i = blockIdx.x * blockDim.x + threadIdx.x;
    uint4 z; z.x = 0u; z.y = 0u; z.z = 0u; z.w = 0u;
    for (; i < n4; i += gridDim.x * blockDim.x) p[i] = z;
}

// ---------------------------------------------------------------------------
// Kernel 1: QKV projection (8192 x 2304 x 768), all-bf16 operands, with fused
// dilation-gather scatter.  Block = 256 threads = 8 waves arranged 4(M)x2(N);
// wave computes 32x64 (2x4 WMMA fragments); block covers 128(M) x 128(N).
// Inner loop per K-step: 12 global_load_b128 + 8 v_wmma.
// ---------------------------------------------------------------------------
__global__ __launch_bounds__(256)
void qkv_gemm_scatter(const unsigned short* __restrict__ xb,
                      const unsigned short* __restrict__ wb,
                      const float* __restrict__ qkv_b,
                      unsigned short* __restrict__ qg,
                      unsigned short* __restrict__ kg,
                      unsigned short* __restrict__ vt) {
    const int lane  = threadIdx.x & 31;
    const int wid   = threadIdx.x >> 5;
    const int row16 = lane & 15;
    const int half  = lane >> 4;
    const int mb = blockIdx.y * 128 + (wid & 3) * 32;
    const int nb = blockIdx.x * 128 + (wid >> 2) * 64;

    f32x8 acc[2][4] = {};

    for (int kk = 0; kk < EMBED_DIM; kk += 32) {
        BFrag a[2], b[4];
#pragma unroll
        for (int s = 0; s < 2; ++s) {
            // A fragment: u[0..3] at element kk+8*half, u[4..7] 32B later.
            const uint4* ap = (const uint4*)(xb +
                (size_t)(mb + s * 16 + row16) * EMBED_DIM + kk + 8 * half);
            a[s].q[0] = ap[0];
            a[s].q[1] = ap[2];
        }
#pragma unroll
        for (int t = 0; t < 4; ++t) {
            // B fragment: u[0..3] at element kk+8*half, u[4..7] 32B later.
            const uint4* bp = (const uint4*)(wb +
                (size_t)(nb + t * 16 + row16) * EMBED_DIM + kk + 8 * half);
            b[t].q[0] = bp[0];
            b[t].q[1] = bp[2];
        }
#pragma unroll
        for (int sm = 0; sm < 2; ++sm)
#pragma unroll
            for (int sn = 0; sn < 4; ++sn)
                acc[sm][sn] = __builtin_amdgcn_wmma_f32_16x16x32_bf16(
                    false, a[sm].v, false, b[sn].v, (short)0, acc[sm][sn],
                    false, false);
    }

    // Epilogue: bias + scatter into gathered layouts.
#pragma unroll
    for (int sm = 0; sm < 2; ++sm) {
#pragma unroll
        for (int sn = 0; sn < 4; ++sn) {
            const int n     = nb + sn * 16 + row16;      // 0..2303
            const int comp  = n / EMBED_DIM;             // 0=Q 1=K 2=V
            const int rem   = n - comp * EMBED_DIM;
            const int h     = rem >> 6;                  // head
            const int d     = rem & 63;
            const int shift = h >> 2;                    // log2(r)
            const int r     = 1 << shift;
            const int off   = (h & 3) & (r - 1);
            const int hb    = head_base_rows(h);
            const float bias = qkv_b[n];
#pragma unroll
            for (int v = 0; v < 8; ++v) {
                const int tok = mb + sm * 16 + half * 8 + v;   // token index
                if ((tok & (r - 1)) != off) continue;          // head does not own
                const float val = acc[sm][sn][v] + bias;
                const int wseg = 2048 << shift;
                const int seg  = tok >> (11 + shift);
                const int tloc = ((tok & (wseg - 1)) - off) >> shift;
                const unsigned short bv = f32_to_bf16u(val);
                if (comp == 0) {
                    qg[(size_t)(hb + seg * 2048 + tloc) * 64 + d] = bv;
                } else if (comp == 1) {
                    kg[(size_t)(hb + seg * 2048 + tloc) * 64 + d] = bv;
                } else {
                    vt[(size_t)hb * 64 +
                       ((size_t)(seg * 64 + d)) * 2048 + tloc] = bv;
                }
            }
        }
    }
}

// ---------------------------------------------------------------------------
// Kernel 2: flash attention over 28 dense (2048 x 2048 x 64) problems.
// Block = 128 threads = 4 waves; each wave owns 16 query rows; grid =
// (2048/64 q-tiles, 28 problems). Online softmax; P converted D->A layout
// through a per-wave LDS strip.  8 WMMAs per 32-key chunk.
// ---------------------------------------------------------------------------
__global__ __launch_bounds__(128)
void dilated_attention(const unsigned short* __restrict__ qg,
                       const unsigned short* __restrict__ kg,
                       const unsigned short* __restrict__ vt,
                       unsigned short* __restrict__ attn) {
    __shared__ __align__(16) unsigned short plds[4][16][32];

    const int lane  = threadIdx.x & 31;
    const int wid   = threadIdx.x >> 5;
    const int row16 = lane & 15;
    const int half  = lane >> 4;

    // problem id -> (head, segment)
    const int pid = blockIdx.y;
    int h, seg;
    if (pid < 16)      { h = pid >> 2;              seg = pid & 3; }
    else if (pid < 24) { h = 4 + ((pid - 16) >> 1); seg = (pid - 16) & 1; }
    else               { h = 8 + (pid - 24);        seg = 0; }

    const int shift = h >> 2;
    const int off   = (h & 3) & ((1 << shift) - 1);
    const int hb    = head_base_rows(h);
    const size_t qkbase = (size_t)(hb + seg * 2048) * 64;
    const size_t vtbase = (size_t)hb * 64 + (size_t)seg * 64 * 2048;

    const int q0 = blockIdx.x * 64 + wid * 16;   // query base within problem

    // Load Q A-fragments (d = 0..63 -> two 16x32 fragments), kept in VGPRs.
    BFrag qf[2];
    {
        const uint4* qp = (const uint4*)(qg + qkbase +
                                         (size_t)(q0 + row16) * 64);
#pragma unroll
        for (int f = 0; f < 2; ++f) {
            qf[f].q[0] = qp[4 * f + half];
            qf[f].q[1] = qp[4 * f + half + 2];
        }
    }

    const float cscale = 0.125f * 1.4426950408889634f;   // d^-0.5 * log2(e)

    f32x8 oacc[4] = {};
    float mrun[8], srun[8];
#pragma unroll
    for (int v = 0; v < 8; ++v) { mrun[v] = -1e30f; srun[v] = 0.0f; }

    for (int kc = 0; kc < 2048; kc += 32) {
        // ---- S = Q K^T for a 16 x 32 score tile (2 key sub-tiles) ----
        f32x8 sacc[2] = {};
#pragma unroll
        for (int kt = 0; kt < 2; ++kt) {
            const uint4* kp = (const uint4*)(kg + qkbase +
                (size_t)(kc + kt * 16 + row16) * 64);
            BFrag bf0, bf1;
            bf0.q[0] = kp[half];
            bf0.q[1] = kp[2 + half];
            bf1.q[0] = kp[4 + half];
            bf1.q[1] = kp[6 + half];
            sacc[kt] = __builtin_amdgcn_wmma_f32_16x16x32_bf16(
                false, qf[0].v, false, bf0.v, (short)0, sacc[kt], false, false);
            sacc[kt] = __builtin_amdgcn_wmma_f32_16x16x32_bf16(
                false, qf[1].v, false, bf1.v, (short)0, sacc[kt], false, false);
        }

        // ---- online softmax: row stats live naturally in the C layout ----
        float alpha[8], rs[8];
#pragma unroll
        for (int v = 0; v < 8; ++v) {
            float mx = fmaxf(sacc[0][v], sacc[1][v]);
#pragma unroll
            for (int msk = 1; msk < 16; msk <<= 1)
                mx = fmaxf(mx, __shfl_xor(mx, msk, 32));
            const float mnew = fmaxf(mrun[v], mx);
            alpha[v] = exp2f((mrun[v] - mnew) * cscale);
            mrun[v]  = mnew;
            const float p0 = exp2f((sacc[0][v] - mnew) * cscale);
            const float p1 = exp2f((sacc[1][v] - mnew) * cscale);
            sacc[0][v] = p0;
            sacc[1][v] = p1;
            float s = p0 + p1;
#pragma unroll
            for (int msk = 1; msk < 16; msk <<= 1)
                s += __shfl_xor(s, msk, 32);
            rs[v] = s;
        }
#pragma unroll
        for (int v = 0; v < 8; ++v) {
            srun[v] = srun[v] * alpha[v] + rs[v];
#pragma unroll
            for (int dt = 0; dt < 4; ++dt) oacc[dt][v] *= alpha[v];
        }

        // ---- P: D-layout -> A-layout via private LDS strip ----
#pragma unroll
        for (int kt = 0; kt < 2; ++kt)
#pragma unroll
            for (int v = 0; v < 8; ++v)
                plds[wid][half * 8 + v][kt * 16 + row16] =
                    f32_to_bf16u(sacc[kt][v]);
        __syncthreads();

        BFrag pf;
        {
            const uint4* prow = (const uint4*)&plds[wid][row16][0];
            pf.q[0] = prow[half];
            pf.q[1] = prow[2 + half];
        }

        // ---- O += P V  (4 d-tiles; Vt pairs contiguous along key dim) ----
#pragma unroll
        for (int dt = 0; dt < 4; ++dt) {
            const uint4* vp = (const uint4*)(vt + vtbase +
                (size_t)(dt * 16 + row16) * 2048 + kc);
            BFrag bv;
            bv.q[0] = vp[half];
            bv.q[1] = vp[2 + half];
            oacc[dt] = __builtin_amdgcn_wmma_f32_16x16x32_bf16(
                false, pf.v, false, bv.v, (short)0, oacc[dt], false, false);
        }
        __syncthreads();
    }

    // ---- normalize and scatter to full-sequence attn buffer (bf16) ----
    const int wseg = 2048 << shift;
#pragma unroll
    for (int dt = 0; dt < 4; ++dt) {
        const int d = dt * 16 + row16;
#pragma unroll
        for (int v = 0; v < 8; ++v) {
            const int tq    = q0 + half * 8 + v;
            const int token = seg * wseg + off + (tq << shift);
            const float o   = oacc[dt][v] / srun[v];
            attn[(size_t)token * EMBED_DIM + h * 64 + d] = f32_to_bf16u(o);
        }
    }
}

// ---------------------------------------------------------------------------
// Kernel 3: output projection (8192 x 768 x 768), all-bf16 operands.
// Same tiling as kernel 1: wave 32x64, block 128x128.
// ---------------------------------------------------------------------------
__global__ __launch_bounds__(256)
void out_gemm(const unsigned short* __restrict__ attn,
              const unsigned short* __restrict__ wb,
              const float* __restrict__ out_b,
              float* __restrict__ out) {
    const int lane  = threadIdx.x & 31;
    const int wid   = threadIdx.x >> 5;
    const int row16 = lane & 15;
    const int half  = lane >> 4;
    const int mb = blockIdx.y * 128 + (wid & 3) * 32;
    const int nb = blockIdx.x * 128 + (wid >> 2) * 64;

    f32x8 acc[2][4] = {};

    for (int kk = 0; kk < EMBED_DIM; kk += 32) {
        BFrag a[2], b[4];
#pragma unroll
        for (int s = 0; s < 2; ++s) {
            const uint4* ap = (const uint4*)(attn +
                (size_t)(mb + s * 16 + row16) * EMBED_DIM + kk + 8 * half);
            a[s].q[0] = ap[0];
            a[s].q[1] = ap[2];
        }
#pragma unroll
        for (int t = 0; t < 4; ++t) {
            const uint4* bp = (const uint4*)(wb +
                (size_t)(nb + t * 16 + row16) * EMBED_DIM + kk + 8 * half);
            b[t].q[0] = bp[0];
            b[t].q[1] = bp[2];
        }
#pragma unroll
        for (int sm = 0; sm < 2; ++sm)
#pragma unroll
            for (int sn = 0; sn < 4; ++sn)
                acc[sm][sn] = __builtin_amdgcn_wmma_f32_16x16x32_bf16(
                    false, a[sm].v, false, b[sn].v, (short)0, acc[sm][sn],
                    false, false);
    }

#pragma unroll
    for (int sm = 0; sm < 2; ++sm) {
#pragma unroll
        for (int sn = 0; sn < 4; ++sn) {
            const int n = nb + sn * 16 + row16;
            const float bias = out_b[n];
#pragma unroll
            for (int v = 0; v < 8; ++v) {
                const int m = mb + sm * 16 + half * 8 + v;
                out[(size_t)m * EMBED_DIM + n] = acc[sm][sn][v] + bias;
            }
        }
    }
}

// ---------------------------------------------------------------------------
// Host launcher
// ---------------------------------------------------------------------------
extern "C" void kernel_launch(void* const* d_in, const int* in_sizes, int n_in,
                              void* d_out, int out_size, void* d_ws,
                              size_t ws_size, hipStream_t stream) {
    (void)in_sizes; (void)n_in; (void)out_size; (void)ws_size;

    const float* x     = (const float*)d_in[0];
    const float* qkv_w = (const float*)d_in[1];
    const float* qkv_b = (const float*)d_in[2];
    const float* out_w = (const float*)d_in[3];
    const float* out_b = (const float*)d_in[4];
    float*       out   = (float*)d_out;

    char* ws = (char*)d_ws;
    unsigned short* qg    = (unsigned short*)(ws);
    unsigned short* kg    = (unsigned short*)(ws + KG_OFF);
    unsigned short* vt    = (unsigned short*)(ws + VT_OFF);
    unsigned short* attn  = (unsigned short*)(ws + ATTN_OFF);
    unsigned short* xb    = (unsigned short*)(ws + XB_OFF);
    unsigned short* wqkvb = (unsigned short*)(ws + WQKV_OFF);
    unsigned short* woutb = (unsigned short*)(ws + WOUT_OFF);

    // One-shot fp32 -> bf16 conversions (hoists conversion off GEMM hot loops).
    cvt_f32_bf16<<<512, 256, 0, stream>>>((const float4*)x, (uint2*)xb,
                                          SEQ_LEN * EMBED_DIM / 4);
    cvt_f32_bf16<<<256, 256, 0, stream>>>((const float4*)qkv_w, (uint2*)wqkvb,
                                          3 * EMBED_DIM * EMBED_DIM / 4);
    cvt_f32_bf16<<<128, 256, 0, stream>>>((const float4*)out_w, (uint2*)woutb,
                                          EMBED_DIM * EMBED_DIM / 4);

    // attn buffer must start at zero every call (dilated holes stay zero).
    zero_attn_kernel<<<1024, 256, 0, stream>>>((uint4*)attn,
                                               (int)(ATTN_BYTES / 16));

    // QKV projection + gather-scatter. grid: (2304/128 N-tiles, 8192/128 M-tiles)
    qkv_gemm_scatter<<<dim3(18, 64), 256, 0, stream>>>(xb, wqkvb, qkv_b,
                                                       qg, kg, vt);

    // 28 attention problems x 32 query tiles of 64 rows.
    dilated_attention<<<dim3(32, 28), 128, 0, stream>>>(qg, kg, vt, attn);

    // Output projection. grid: (768/128 N-tiles, 8192/128 M-tiles)
    out_gemm<<<dim3(6, 64), 256, 0, stream>>>(attn, woutb, out_b, out);
}
